// MaskedAttentionHead_9242769622267
// MI455X (gfx1250) — matrix operand order repaired and checked by reading, not compiled, not run on hardware
//
#include <hip/hip_runtime.h>
#include <hip/hip_bf16.h>

// ---------------------------------------------------------------------------
// MaskedAttentionHead for MI455X (gfx1250, wave32, WMMA 16x16x32 f16)
// B=32, S=1024, D=768, HS=64
// ---------------------------------------------------------------------------

#define B_   32
#define S_   1024
#define D_   768
#define HS_  64
#define EPS_ 1e-5f
#define LOG2E_ 1.44269504088896340736f

typedef __attribute__((ext_vector_type(16))) _Float16      v16h;
typedef __attribute__((ext_vector_type(8)))  float         v8f;
typedef __attribute__((ext_vector_type(4)))  unsigned int  u32x4;
typedef __attribute__((ext_vector_type(4)))  _Float16      h4;
typedef __attribute__((ext_vector_type(8)))  _Float16      h8;

union AB  { v16h v; u32x4 q[2]; };   // 32B WMMA operand built from two 16B runs
union H8U { h8   h; u32x4 q;    };   // 16B packed f16 store

// ---------------------------------------------------------------------------
// Kernel 0: pack Wq(*0.125), Wk, Wv -> f16 Wall[192][768]
// rows 0..63 = Q (scale 1/sqrt(64) folded), 64..127 = K, 128..191 = V
// ---------------------------------------------------------------------------
__global__ void prep_weights(const float* __restrict__ Wk,
                             const float* __restrict__ Wq,
                             const float* __restrict__ Wv,
                             _Float16* __restrict__ Wall) {
    int i = blockIdx.x * 256 + threadIdx.x;          // < 192*768
    int h = i / D_, d = i - h * D_;
    float v;
    if (h < 64)        v = Wq[h * D_ + d] * 0.125f;
    else if (h < 128)  v = Wk[(h - 64) * D_ + d];
    else               v = Wv[(h - 128) * D_ + d];
    Wall[i] = (_Float16)v;
}

// ---------------------------------------------------------------------------
// Kernel 1: LayerNorm (wave per row) + QKV projection via WMMA
// 32 rows / block, 256 threads (8 waves). K = 768 -> 24 wmma k-steps.
// qbuf,kbuf: f16 [B*S][64]   vT: f16 [B][64][S]
// ---------------------------------------------------------------------------
__global__ void __launch_bounds__(256)
ln_qkv(const float* __restrict__ x,
       const float* __restrict__ gamma,
       const float* __restrict__ beta,
       const _Float16* __restrict__ Wall,
       _Float16* __restrict__ qbuf,
       _Float16* __restrict__ kbuf,
       _Float16* __restrict__ vT) {
    __shared__ alignas(16) _Float16 xn[32][D_ + 8];   // padded stride 776 halfs

    const int tid  = threadIdx.x;
    const int wave = tid >> 5;
    const int lane = tid & 31;
    const int row0 = blockIdx.x * 32;                 // global token base

    // ---- Stage 1: LayerNorm, write f16 normalized rows to LDS ----
    const float4* g4 = (const float4*)gamma;
    const float4* b4 = (const float4*)beta;
    #pragma unroll
    for (int i = 0; i < 4; ++i) {
        const int row = wave * 4 + i;
        const float* xr = x + (size_t)(row0 + row) * D_;
        float4 xv[6];
        float s = 0.f, sq = 0.f;
        #pragma unroll
        for (int t = 0; t < 6; ++t) {
            xv[t] = *(const float4*)(xr + lane * 4 + t * 128);
            s  += xv[t].x + xv[t].y + xv[t].z + xv[t].w;
            sq += xv[t].x * xv[t].x + xv[t].y * xv[t].y +
                  xv[t].z * xv[t].z + xv[t].w * xv[t].w;
        }
        #pragma unroll
        for (int m = 1; m < 32; m <<= 1) {
            s  += __shfl_xor(s,  m);
            sq += __shfl_xor(sq, m);
        }
        const float mean = s * (1.0f / D_);
        const float var  = sq * (1.0f / D_) - mean * mean;
        const float rstd = rsqrtf(var + EPS_);
        #pragma unroll
        for (int t = 0; t < 6; ++t) {
            const int d  = lane * 4 + t * 128;
            const float4 g = g4[lane + t * 32];
            const float4 bb = b4[lane + t * 32];
            h4 o;
            o[0] = (_Float16)((xv[t].x - mean) * rstd * g.x + bb.x);
            o[1] = (_Float16)((xv[t].y - mean) * rstd * g.y + bb.y);
            o[2] = (_Float16)((xv[t].z - mean) * rstd * g.z + bb.z);
            o[3] = (_Float16)((xv[t].w - mean) * rstd * g.w + bb.w);
            *(h4*)&xn[row][d] = o;
        }
    }
    __syncthreads();

    // ---- Stage 2: GEMM xn(32x768) * Wall^T(768x192) via WMMA ----
    const int half16 = (lane < 16) ? 0 : 1;
    const int arow16 = lane & 15;
    const int a0 = half16 ? 16 : 0;       // A-layout K-run starts: {0,8} / {16,24}

    #pragma unroll
    for (int it = 0; it < 3; ++it) {
        const int t    = wave + 8 * it;   // 24 tiles: 2 row-tiles x 12 col-tiles
        const int mrow = (t & 1) * 16;
        const int ncol = (t >> 1) * 16;

        v8f acc = {};
        const _Float16* arow_p = &xn[mrow + arow16][0];
        const int h  = ncol + arow16;
        const int db = half16 ? 16 : 0;
        const _Float16* brow_p = Wall + h * D_;

        #pragma unroll 4
        for (int kb = 0; kb < 24; ++kb) {
            const int d = kb * 32;
            AB a;
            a.q[0] = *(const u32x4*)(arow_p + d + a0);
            a.q[1] = *(const u32x4*)(arow_p + d + a0 + 8);
            v16h bv = *(const v16h*)(brow_p + d + db);
            acc = __builtin_amdgcn_wmma_f32_16x16x32_f16(
                      false, a.v, false, bv, (short)0, acc, false, false);
        }

        const int col   = ncol + arow16;
        const int rbase = row0 + mrow + (half16 ? 8 : 0);
        if (col < 64) {
            #pragma unroll
            for (int r = 0; r < 8; ++r)
                qbuf[(size_t)(rbase + r) * HS_ + col] = (_Float16)acc[r];
        } else if (col < 128) {
            #pragma unroll
            for (int r = 0; r < 8; ++r)
                kbuf[(size_t)(rbase + r) * HS_ + (col - 64)] = (_Float16)acc[r];
        } else {
            // V stored transposed: vT[b][h][s], 8 consecutive s -> one 16B store
            const int hh    = col - 128;
            const int bb    = row0 >> 10;
            const int sbase = (row0 & (S_ - 1)) + mrow + (half16 ? 8 : 0);
            H8U o;
            #pragma unroll
            for (int r = 0; r < 8; ++r) o.h[r] = (_Float16)acc[r];
            *(h8*)(vT + ((size_t)(bb * HS_ + hh) << 10) + sbase) = o.h;
        }
    }
}

// ---------------------------------------------------------------------------
// Kernel 2: causal flash attention. 8 waves x 16 queries / block, 64-key steps.
// QK^T, P*V and the softmax row-sum all on v_wmma_f32_16x16x32_f16.
// P converted C->A layout through a per-wave LDS tile.
// ---------------------------------------------------------------------------
__global__ void __launch_bounds__(256)
attn(const _Float16* __restrict__ qbuf,
     const _Float16* __restrict__ kbuf,
     const _Float16* __restrict__ vT,
     float* __restrict__ out) {
    __shared__ alignas(16) _Float16 plds[8][16][72];  // per-wave 16x64 P tile (padded)

    const int tid  = threadIdx.x;
    const int wave = tid >> 5;
    const int lane = tid & 31;

    const int qg = blockIdx.x * 128 + wave * 16;      // first global query token
    const int b  = qg >> 10;
    const int ql = qg & (S_ - 1);                     // query index within batch

    const int half16 = (lane < 16) ? 0 : 1;
    const int colj   = lane & 15;
    const int a0     = half16 ? 16 : 0;

    // Q tile in A layout (16 queries x 64 dims -> two 32-K chunks)
    AB qa[2];
    #pragma unroll
    for (int kk = 0; kk < 2; ++kk) {
        const _Float16* qp = qbuf + (size_t)(qg + colj) * HS_ + kk * 32;
        qa[kk].q[0] = *(const u32x4*)(qp + a0);
        qa[kk].q[1] = *(const u32x4*)(qp + a0 + 8);
    }

    // all-ones B operand (64x16 ones in two 32x16 chunks -> same vector)
    v16h ones;
    #pragma unroll
    for (int i = 0; i < 16; ++i) ones[i] = (_Float16)1.0f;

    float m[8], l[8], al[8];
    v8f o[4];
    #pragma unroll
    for (int r = 0; r < 8; ++r) { m[r] = -1e30f; l[r] = 0.f; }
    #pragma unroll
    for (int nt = 0; nt < 4; ++nt) o[nt] = (v8f){};

    const int nblk = (ql + 79) >> 6;                  // causal: keys 0..ql+15
    const int rq0  = ql + (half16 ? 8 : 0);

    for (int jb = 0; jb < nblk; ++jb) {
        const int j0 = jb * 64;

        if (jb + 1 < nblk) {  // prefetch next K/V block -> global_prefetch
            __builtin_prefetch(kbuf + (size_t)(b * S_ + j0 + 64 + colj * 4) * HS_, 0, 1);
            __builtin_prefetch(vT + ((size_t)(b * HS_ + colj * 4) << 10) + j0 + 64, 0, 1);
        }

        // ---- scores: four 16x16 tiles, accumulate over d = 0..63 ----
        v8f sc[4];
        #pragma unroll
        for (int jt = 0; jt < 4; ++jt) {
            const _Float16* kp =
                kbuf + (size_t)(b * S_ + j0 + jt * 16 + colj) * HS_ + (half16 ? 16 : 0);
            v16h b0 = *(const v16h*)(kp);
            v16h b1 = *(const v16h*)(kp + 32);
            v8f a = {};
            a = __builtin_amdgcn_wmma_f32_16x16x32_f16(
                    false, qa[0].v, false, b0, (short)0, a, false, false);
            a = __builtin_amdgcn_wmma_f32_16x16x32_f16(
                    false, qa[1].v, false, b1, (short)0, a, false, false);
            sc[jt] = a;
        }

        // ---- causal mask + online softmax (max reduction in 16-lane halves) ----
        #pragma unroll
        for (int r = 0; r < 8; ++r) {
            const int q = rq0 + r;
            float s0 = (j0 + colj      <= q) ? sc[0][r] : -1e30f;
            float s1 = (j0 + 16 + colj <= q) ? sc[1][r] : -1e30f;
            float s2 = (j0 + 32 + colj <= q) ? sc[2][r] : -1e30f;
            float s3 = (j0 + 48 + colj <= q) ? sc[3][r] : -1e30f;

            float t = fmaxf(fmaxf(s0, s1), fmaxf(s2, s3));
            #pragma unroll
            for (int mk = 1; mk < 16; mk <<= 1) t = fmaxf(t, __shfl_xor(t, mk));
            const float mn = fmaxf(m[r], t);

            al[r] = __builtin_exp2f((m[r] - mn) * LOG2E_);
            m[r]  = mn;

            const float p0 = __builtin_exp2f((s0 - mn) * LOG2E_);
            const float p1 = __builtin_exp2f((s1 - mn) * LOG2E_);
            const float p2 = __builtin_exp2f((s2 - mn) * LOG2E_);
            const float p3 = __builtin_exp2f((s3 - mn) * LOG2E_);

            const int prow = r + (half16 ? 8 : 0);
            plds[wave][prow][colj]      = (_Float16)p0;
            plds[wave][prow][16 + colj] = (_Float16)p1;
            plds[wave][prow][32 + colj] = (_Float16)p2;
            plds[wave][prow][48 + colj] = (_Float16)p3;

            #pragma unroll
            for (int nt = 0; nt < 4; ++nt) o[nt][r] *= al[r];
        }

        asm volatile("s_wait_dscnt 0x0" ::: "memory");  // wave-local LDS transpose

        // ---- reload P in A layout (two 16x32 chunks) ----
        AB pa[2];
        const _Float16* pp = &plds[wave][colj][0];
        pa[0].q[0] = *(const u32x4*)(pp + a0);
        pa[0].q[1] = *(const u32x4*)(pp + a0 + 8);
        pa[1].q[0] = *(const u32x4*)(pp + 32 + a0);
        pa[1].q[1] = *(const u32x4*)(pp + 32 + a0 + 8);

        // ---- row sums of P via WMMA with all-ones B (replaces shfl tree) ----
        v8f ps = {};
        ps = __builtin_amdgcn_wmma_f32_16x16x32_f16(
                 false, pa[0].v, false, ones, (short)0, ps, false, false);
        ps = __builtin_amdgcn_wmma_f32_16x16x32_f16(
                 false, pa[1].v, false, ones, (short)0, ps, false, false);
        #pragma unroll
        for (int r = 0; r < 8; ++r) l[r] = l[r] * al[r] + ps[r];

        // ---- P (A layout) x V (B layout from vT) ----
        #pragma unroll
        for (int nt = 0; nt < 4; ++nt) {
            const _Float16* vp =
                vT + ((size_t)(b * HS_ + nt * 16 + colj) << 10) + j0 + (half16 ? 16 : 0);
            v16h vb0 = *(const v16h*)(vp);
            v16h vb1 = *(const v16h*)(vp + 32);
            o[nt] = __builtin_amdgcn_wmma_f32_16x16x32_f16(
                        false, pa[0].v, false, vb0, (short)0, o[nt], false, false);
            o[nt] = __builtin_amdgcn_wmma_f32_16x16x32_f16(
                        false, pa[1].v, false, vb1, (short)0, o[nt], false, false);
        }
    }

    // ---- epilogue: normalize and store f32 output [token][64] ----
    float inv[8];
    #pragma unroll
    for (int r = 0; r < 8; ++r) inv[r] = 1.0f / l[r];
    #pragma unroll
    for (int nt = 0; nt < 4; ++nt) {
        #pragma unroll
        for (int r = 0; r < 8; ++r) {
            const int tok = qg + r + (half16 ? 8 : 0);
            out[(size_t)tok * HS_ + nt * 16 + colj] = o[nt][r] * inv[r];
        }
    }
}

// ---------------------------------------------------------------------------
// Launch
// ---------------------------------------------------------------------------
extern "C" void kernel_launch(void* const* d_in, const int* in_sizes, int n_in,
                              void* d_out, int out_size, void* d_ws, size_t ws_size,
                              hipStream_t stream) {
    const float* x     = (const float*)d_in[0];
    const float* gamma = (const float*)d_in[1];
    const float* beta  = (const float*)d_in[2];
    const float* Wk    = (const float*)d_in[3];
    const float* Wq    = (const float*)d_in[4];
    const float* Wv    = (const float*)d_in[5];
    float* out = (float*)d_out;

    char* ws = (char*)d_ws;
    const size_t WALL_B = (size_t)192 * D_ * sizeof(_Float16);      // 294912
    const size_t QKV_B  = (size_t)B_ * S_ * HS_ * sizeof(_Float16); // 4194304
    _Float16* Wall = (_Float16*)ws;
    _Float16* qbuf = (_Float16*)(ws + WALL_B);
    _Float16* kbuf = (_Float16*)(ws + WALL_B + QKV_B);
    _Float16* vT   = (_Float16*)(ws + WALL_B + 2 * QKV_B);

    prep_weights<<<(192 * D_) / 256, 256, 0, stream>>>(Wk, Wq, Wv, Wall);
    ln_qkv<<<(B_ * S_) / 32, 256, 0, stream>>>(x, gamma, beta, Wall, qbuf, kbuf, vT);
    attn<<<(B_ * S_) / 128, 256, 0, stream>>>(qbuf, kbuf, vT, out);
}
